// MultiHeadAttention_51616916964161
// MI455X (gfx1250) — matrix operand order repaired and checked by reading, not compiled, not run on hardware
//
#include <hip/hip_runtime.h>

#define DIMN   1024
#define HEADS  16
#define HDM    64
#define BATCH  2
#define SEQ    2048
#define MTOT   (BATCH * SEQ)

typedef __attribute__((ext_vector_type(16))) __bf16 bf16x16;
typedef __attribute__((ext_vector_type(8)))  float  f32x8;

union FragAB {
    bf16x16 v;
    uint4   q[2];
};

__device__ __forceinline__ unsigned short f32_to_bf16(float f) {
    unsigned int u = __float_as_uint(f);
    u += 0x7FFFu + ((u >> 16) & 1u);   // round-to-nearest-even
    return (unsigned short)(u >> 16);
}

__device__ __forceinline__ f32x8 wmma_bf16(const FragAB& a, const FragAB& b, f32x8 c) {
    return __builtin_amdgcn_wmma_f32_16x16x32_bf16(false, a.v, false, b.v,
                                                   (short)0, c, false, false);
}

// CDNA5 async copy: global -> LDS without VGPR round trip (ASYNCcnt tracked).
__device__ __forceinline__ void async_g2l_b128(unsigned ldsOff, unsigned vOff,
                                               const void* base) {
    asm volatile("global_load_async_to_lds_b128 %0, %1, %2"
                 :
                 : "v"(ldsOff), "v"(vOff), "s"(base)
                 : "memory");
}

__device__ __forceinline__ void wait_async0() {
    asm volatile("s_wait_asynccnt 0x0" ::: "memory");
}

__device__ __forceinline__ unsigned lds_off(const void* p) {
    return (unsigned)(uintptr_t)p;
}

// ---------------------------------------------------------------------------
// f32 -> bf16 elementwise
__global__ __launch_bounds__(256) void cvt_bf16_kernel(const float* __restrict__ in,
                                                       unsigned short* __restrict__ out,
                                                       int n) {
    int i = blockIdx.x * blockDim.x + threadIdx.x;
    if (i < n) out[i] = f32_to_bf16(in[i]);
}

// f32 [K,N] -> bf16 transposed [N,K]
__global__ __launch_bounds__(256) void cvt_wT_kernel(const float* __restrict__ w,
                                                     unsigned short* __restrict__ wT,
                                                     int K, int N) {
    int i = blockIdx.x * blockDim.x + threadIdx.x;
    if (i < K * N) {
        int k = i / N, n = i % N;
        wT[(size_t)n * K + k] = f32_to_bf16(w[i]);
    }
}

// ---------------------------------------------------------------------------
// C[M,N] = A[M,K] * B[K,N] + bias; A bf16 row-major, BT bf16 [N,K].
// Block tile 128x64, 4 waves, each wave 32x64 (8 WMMA / K-step).
// mode 0: write bf16 head layout [B,H,S,HD]
// mode 1: write f32 flat [M,N]
// mode 2: write bf16 transposed head layout [B,H,HD,S]  (for V)
#define LDT 40   // LDS row stride (elements); 80 bytes, 16B aligned

__global__ __launch_bounds__(128) void gemm_bias_kernel(
        const unsigned short* __restrict__ A,
        const unsigned short* __restrict__ BT,
        const float* __restrict__ bias,
        unsigned short* __restrict__ out_bf16,
        float* __restrict__ out_f32,
        int M, int N, int K, int mode) {
    __shared__ unsigned short lA[128 * LDT];
    __shared__ unsigned short lB[64 * LDT];

    const int tid  = threadIdx.x;
    const int lane = tid & 31;
    const int w    = tid >> 5;
    const int m0   = blockIdx.y * 128;
    const int n0   = blockIdx.x * 64;

    const int mi   = lane & 15;
    const int kh8  = (lane >> 4) * 8;    // A-operand K split
    const int kb16 = (lane >> 4) * 16;   // unused (B operand also uses kh8)

    f32x8 acc[2][4];
    for (int a = 0; a < 2; ++a)
        for (int t = 0; t < 4; ++t)
            for (int i = 0; i < 8; ++i) acc[a][t][i] = 0.f;

    // branch-free per-thread chunk coordinates (16B chunks of a 32-wide K slab)
    const int rA  = tid >> 2;            // base A row for j-loop (step 32 rows)
    const int ckA = (tid & 3) * 8;

    for (int kk = 0; kk < K; kk += 32) {
        // A tile: 128 rows x 32 cols = 512 chunks, 4 per thread
        for (int j = 0; j < 4; ++j) {
            int r = rA + 32 * j;
            async_g2l_b128(lds_off(&lA[r * LDT + ckA]),
                           (unsigned)(((m0 + r) * K + kk + ckA) * 2), A);
        }
        // B tile: 64 rows x 32 cols = 256 chunks, 2 per thread
        for (int j = 0; j < 2; ++j) {
            int r = rA + 32 * j;
            async_g2l_b128(lds_off(&lB[r * LDT + ckA]),
                           (unsigned)(((n0 + r) * K + kk + ckA) * 2), BT);
        }
        if (kk + 32 < K) {
            __builtin_prefetch(&A [(size_t)(m0 + tid) * K + kk + 32], 0, 1);
            __builtin_prefetch(&BT[(size_t)(n0 + (tid >> 1)) * K + kk + 32], 0, 1);
        }
        wait_async0();
        __syncthreads();

        FragAB fa[2];
        for (int a = 0; a < 2; ++a) {
            const unsigned short* arow = &lA[(w * 32 + a * 16 + mi) * LDT];
            fa[a].q[0] = *(const uint4*)&arow[kh8];
            fa[a].q[1] = *(const uint4*)&arow[kh8 + 16];
        }
        for (int t = 0; t < 4; ++t) {
            FragAB fb;
            fb.q[0] = *(const uint4*)&lB[(t * 16 + mi) * LDT + kh8];
            fb.q[1] = *(const uint4*)&lB[(t * 16 + mi) * LDT + kh8 + 16];
            acc[0][t] = wmma_bf16(fa[0], fb, acc[0][t]);
            acc[1][t] = wmma_bf16(fa[1], fb, acc[1][t]);
        }
        __syncthreads();
    }

    const int mrow = (lane >> 4) * 8;
    for (int a = 0; a < 2; ++a) {
        for (int t = 0; t < 4; ++t) {
            int col = n0 + t * 16 + mi;
            float bcol = bias[col];
            for (int vv = 0; vv < 8; ++vv) {
                int row = m0 + w * 32 + a * 16 + mrow + vv;
                float val = acc[a][t][vv] + bcol;
                if (mode == 0) {
                    int bb = row / SEQ, ss = row % SEQ;
                    int hh = col / HDM, dd = col % HDM;
                    out_bf16[(((size_t)bb * HEADS + hh) * SEQ + ss) * HDM + dd] =
                        f32_to_bf16(val);
                } else if (mode == 2) {
                    int bb = row / SEQ, ss = row % SEQ;
                    int hh = col / HDM, dd = col % HDM;
                    out_bf16[(((size_t)bb * HEADS + hh) * HDM + dd) * SEQ + ss] =
                        f32_to_bf16(val);
                } else {
                    out_f32[(size_t)row * N + col] = val;
                }
            }
        }
    }
}

// ---------------------------------------------------------------------------
// Flash attention per (b, h, 64-query tile). 4 waves, 16 query rows each.
// K tile [key][d] and V tile [d][key] both staged with async-to-LDS
// (V is pre-transposed to [B,H,HD,S] by the V projection, mode 2).
#define LKT 72   // LDS row stride (144 bytes, 16B aligned)

__global__ __launch_bounds__(128) void flash_attn_kernel(
        const unsigned short* __restrict__ qh,
        const unsigned short* __restrict__ kh,
        const unsigned short* __restrict__ vhT,
        unsigned short* __restrict__ ctx) {   // [B,S,DIM] bf16
    __shared__ unsigned short lK[64 * LKT];        // [key][d]
    __shared__ unsigned short lV[64 * LKT];        // [d][key]
    __shared__ unsigned short lP[4][16 * LKT];     // per-wave P tile

    const int tid  = threadIdx.x;
    const int lane = tid & 31;
    const int w    = tid >> 5;
    const int b    = blockIdx.z;
    const int h    = blockIdx.y;
    const int q0   = blockIdx.x * 64;

    const int mi   = lane & 15;
    const int kh8  = (lane >> 4) * 8;
    const int kb16 = (lane >> 4) * 16;
    const int mrow = (lane >> 4) * 8;

    const size_t headBase = ((size_t)b * HEADS + h) * SEQ;
    const unsigned short* kbase = kh  + headBase * HDM;            // [S,HD]
    const unsigned short* vbase = vhT + headBase * HDM;            // [HD,S]

    // Q fragments (A operand) for rows q0 + w*16 + mi
    FragAB aq0, aq1;
    {
        const unsigned short* qrow = qh + (headBase + q0 + w * 16 + mi) * HDM;
        aq0.q[0] = *(const uint4*)&qrow[kh8];
        aq0.q[1] = *(const uint4*)&qrow[kh8 + 16];
        aq1.q[0] = *(const uint4*)&qrow[32 + kh8];
        aq1.q[1] = *(const uint4*)&qrow[32 + kh8 + 16];
    }

    float rowmax[8], rowsum[8];
    f32x8 accO[4];
    for (int vv = 0; vv < 8; ++vv) { rowmax[vv] = -1e30f; rowsum[vv] = 0.f; }
    for (int t = 0; t < 4; ++t)
        for (int vv = 0; vv < 8; ++vv) accO[t][vv] = 0.f;

    unsigned short* lPw = lP[w];

    const int rT = tid >> 3;             // tile row 0..15 (base, step 16)
    const int cT = (tid & 7) * 8;        // tile col chunk

    for (int kb = 0; kb < SEQ; kb += 64) {
        // stage K [key][d] and V [d][key]: 64x64 each, 16B async chunks
        for (int j = 0; j < 4; ++j) {
            int r = rT + 16 * j;
            async_g2l_b128(lds_off(&lK[r * LKT + cT]),
                           (unsigned)(((kb + r) * HDM + cT) * 2), kbase);
            async_g2l_b128(lds_off(&lV[r * LKT + cT]),
                           (unsigned)((r * SEQ + kb + cT) * 2), vbase);
        }
        wait_async0();
        __syncthreads();

        // scores: 4 tiles of 16 keys, K-dim = 64 (2 WMMA each)
        f32x8 sc[4];
        for (int t = 0; t < 4; ++t) {
            for (int vv = 0; vv < 8; ++vv) sc[t][vv] = 0.f;
            const unsigned short* krow = &lK[(t * 16 + mi) * LKT];
            FragAB bk;
            bk.q[0] = *(const uint4*)&krow[kh8];
            bk.q[1] = *(const uint4*)&krow[kh8 + 16];
            sc[t] = wmma_bf16(aq0, bk, sc[t]);
            bk.q[0] = *(const uint4*)&krow[32 + kh8];
            bk.q[1] = *(const uint4*)&krow[32 + kh8 + 16];
            sc[t] = wmma_bf16(aq1, bk, sc[t]);
        }

        // online softmax (row stats replicated across 16-lane groups)
        for (int vv = 0; vv < 8; ++vv) {
            float m = sc[0][vv];
            for (int t = 1; t < 4; ++t) m = fmaxf(m, sc[t][vv]);
            for (int msk = 1; msk < 16; msk <<= 1) m = fmaxf(m, __shfl_xor(m, msk, 32));
            float nm = fmaxf(rowmax[vv], m);
            float f  = __expf(rowmax[vv] - nm);
            rowmax[vv] = nm;
            float p = 0.f;
            for (int t = 0; t < 4; ++t) {
                sc[t][vv] = __expf(sc[t][vv] - nm);
                p += sc[t][vv];
            }
            for (int msk = 1; msk < 16; msk <<= 1) p += __shfl_xor(p, msk, 32);
            rowsum[vv] = rowsum[vv] * f + p;
            for (int t = 0; t < 4; ++t) accO[t][vv] *= f;
        }

        // redistribute P from C-layout to A-layout via this wave's LDS tile
        for (int t = 0; t < 4; ++t)
            for (int vv = 0; vv < 8; ++vv)
                lPw[(mrow + vv) * LKT + t * 16 + mi] = f32_to_bf16(sc[t][vv]);

        // P @ V : K-dim = 64 keys (2 frags), 4 output d-tiles
        FragAB ap0, ap1;
        const unsigned short* prow = &lPw[mi * LKT];
        ap0.q[0] = *(const uint4*)&prow[kh8];
        ap0.q[1] = *(const uint4*)&prow[kh8 + 16];
        ap1.q[0] = *(const uint4*)&prow[32 + kh8];
        ap1.q[1] = *(const uint4*)&prow[32 + kh8 + 16];
        for (int t = 0; t < 4; ++t) {
            const unsigned short* vrow = &lV[(t * 16 + mi) * LKT];
            FragAB bv;
            bv.q[0] = *(const uint4*)&vrow[kh8];
            bv.q[1] = *(const uint4*)&vrow[kh8 + 16];
            accO[t] = wmma_bf16(ap0, bv, accO[t]);
            bv.q[0] = *(const uint4*)&vrow[32 + kh8];
            bv.q[1] = *(const uint4*)&vrow[32 + kh8 + 16];
            accO[t] = wmma_bf16(ap1, bv, accO[t]);
        }
        __syncthreads();
    }

    // normalize and write context bf16 in [B,S,DIM] (head-concat) layout
    for (int t = 0; t < 4; ++t) {
        int col = h * HDM + t * 16 + mi;
        for (int vv = 0; vv < 8; ++vv) {
            int row = q0 + w * 16 + mrow + vv;
            float val = accO[t][vv] / rowsum[vv];
            ctx[((size_t)b * SEQ + row) * DIMN + col] = f32_to_bf16(val);
        }
    }
}

// ---------------------------------------------------------------------------
extern "C" void kernel_launch(void* const* d_in, const int* in_sizes, int n_in,
                              void* d_out, int out_size, void* d_ws, size_t ws_size,
                              hipStream_t stream) {
    (void)in_sizes; (void)n_in; (void)out_size; (void)ws_size;
    const float* q  = (const float*)d_in[0];
    const float* k  = (const float*)d_in[1];
    const float* v  = (const float*)d_in[2];
    const float* wq = (const float*)d_in[3];
    const float* bq = (const float*)d_in[4];
    const float* wk = (const float*)d_in[5];
    const float* bk = (const float*)d_in[6];
    const float* wv = (const float*)d_in[7];
    const float* bv = (const float*)d_in[8];
    const float* wo = (const float*)d_in[9];
    const float* bo = (const float*)d_in[10];
    float* out = (float*)d_out;

    char* ws = (char*)d_ws;
    size_t off = 0;
    auto take = [&](size_t bytes) {
        char* p = ws + off;
        off += (bytes + 255) & ~(size_t)255;
        return p;
    };

    const size_t actB = (size_t)MTOT * DIMN * 2;   // 8 MB (bf16 activations)
    const size_t wB   = (size_t)DIMN * DIMN * 2;   // 2 MB (bf16 weights)

    unsigned short* xq  = (unsigned short*)take(actB);
    unsigned short* xk  = (unsigned short*)take(actB);
    unsigned short* xv  = (unsigned short*)take(actB);
    unsigned short* wqT = (unsigned short*)take(wB);
    unsigned short* wkT = (unsigned short*)take(wB);
    unsigned short* wvT = (unsigned short*)take(wB);
    unsigned short* woT = (unsigned short*)take(wB);
    unsigned short* qhd = (unsigned short*)take(actB);
    unsigned short* khd = (unsigned short*)take(actB);
    unsigned short* vhdT = (unsigned short*)take(actB);  // [B,H,HD,S]
    unsigned short* ctx = xq;   // reuse: x_q bf16 is dead after the Q projection

    const int nAct = MTOT * DIMN;
    const int nW   = DIMN * DIMN;

    cvt_bf16_kernel<<<nAct / 256, 256, 0, stream>>>(q, xq, nAct);
    cvt_bf16_kernel<<<nAct / 256, 256, 0, stream>>>(k, xk, nAct);
    cvt_bf16_kernel<<<nAct / 256, 256, 0, stream>>>(v, xv, nAct);
    cvt_wT_kernel<<<nW / 256, 256, 0, stream>>>(wq, wqT, DIMN, DIMN);
    cvt_wT_kernel<<<nW / 256, 256, 0, stream>>>(wk, wkT, DIMN, DIMN);
    cvt_wT_kernel<<<nW / 256, 256, 0, stream>>>(wv, wvT, DIMN, DIMN);
    cvt_wT_kernel<<<nW / 256, 256, 0, stream>>>(wo, woT, DIMN, DIMN);

    dim3 ggrid(DIMN / 64, MTOT / 128);
    gemm_bias_kernel<<<ggrid, 128, 0, stream>>>(xq, wqT, bq, qhd, nullptr, MTOT, DIMN, DIMN, 0);
    gemm_bias_kernel<<<ggrid, 128, 0, stream>>>(xk, wkT, bk, khd, nullptr, MTOT, DIMN, DIMN, 0);
    gemm_bias_kernel<<<ggrid, 128, 0, stream>>>(xv, wvT, bv, vhdT, nullptr, MTOT, DIMN, DIMN, 2);

    dim3 agrid(SEQ / 64, HEADS, BATCH);
    flash_attn_kernel<<<agrid, 128, 0, stream>>>(qhd, khd, vhdT, ctx);

    gemm_bias_kernel<<<ggrid, 128, 0, stream>>>(ctx, woT, bo, nullptr, out, MTOT, DIMN, DIMN, 1);
}
